// TFLEDEncoderSelfAttention_55327768707420
// MI455X (gfx1250) — matrix operand order repaired and checked by reading, not compiled, not run on hardware
//
#include <hip/hip_runtime.h>
#include <hip/hip_bf16.h>

// ---------------------------------------------------------------------------
// Problem constants (Longformer self-attention, B=1)
// ---------------------------------------------------------------------------
#define S_LEN  4096
#define E_DIM  768
#define H_NUM  12
#define D_DIM  64
#define W_HALF 256
#define LARGE_NEG (-100000000.0f)

// Attention tiling
#define QB 32        // queries per workgroup
#define WN 576       // padded key window (32 + 2*256 rounded to 64)
#define CH 64        // key chunk
#define NCH (WN / CH)

typedef __bf16 bf16;
typedef bf16  v16bf __attribute__((ext_vector_type(16)));
typedef bf16  v8bf  __attribute__((ext_vector_type(8)));
typedef float v8f   __attribute__((ext_vector_type(8)));

typedef unsigned int u32x4 __attribute__((ext_vector_type(4)));
typedef int          i32x4 __attribute__((ext_vector_type(4)));
typedef int          i32x8 __attribute__((ext_vector_type(8)));

__device__ __forceinline__ v8f wmma_bf16(v16bf a, v16bf b, v8f c) {
  // D = A(16x32 bf16) * B(32x16 bf16) + C(16x16 f32)
  return __builtin_amdgcn_wmma_f32_16x16x32_bf16(
      /*neg_a=*/false, a, /*neg_b=*/false, b,
      /*c_mod=*/(short)0, c, /*reuse_a=*/false, /*reuse_b=*/false);
}

// ---------------------------------------------------------------------------
// TDM: async 2D tile load (bf16 elements) Global -> LDS.
//   Loads tile_rows x 64 bf16 from a row-major [*, 64] tensor starting at
//   gaddr into LDS at lds_addr. Tracked by TENSORcnt (in-order per wave).
// D# packing per CDNA5 ISA (6-arg builtin).
// ---------------------------------------------------------------------------
__device__ __forceinline__ void tdm_load_rows_bf16(const void* gaddr,
                                                   unsigned lds_addr,
                                                   unsigned tile_rows) {
  unsigned long long ga = (unsigned long long)(uintptr_t)gaddr;
  const unsigned tdim0 = D_DIM;       // innermost length (elements)
  const unsigned tdim1 = tile_rows;   // rows available from start
  const unsigned tile0 = D_DIM;       // tile width
  const unsigned tile1 = tile_rows;   // tile height
  const unsigned stride0 = D_DIM;     // row stride (elements)

  u32x4 g0;
  g0[0] = 1u;                                        // count=1 (valid user D#)
  g0[1] = lds_addr;                                  // lds_addr
  g0[2] = (unsigned)(ga & 0xFFFFFFFFu);              // global_addr[31:0]
  g0[3] = (unsigned)((ga >> 32) & 0x1FFFFFFu)        // global_addr[56:32]
          | (2u << 30);                              // type=2 ("image")

  i32x8 g1;
  g1[0] = (int)(1u << 16);                           // data_size=1 (2 bytes)
  g1[1] = (int)((tdim0 & 0xFFFFu) << 16);            // tensor_dim0[15:0]
  g1[2] = (int)(((tdim0 >> 16) & 0xFFFFu) |
                ((tdim1 & 0xFFFFu) << 16));          // dim0 hi | dim1 lo
  g1[3] = (int)(((tdim1 >> 16) & 0xFFFFu) |
                ((tile0 & 0xFFFFu) << 16));          // dim1 hi | tile_dim0
  g1[4] = (int)(tile1 & 0xFFFFu);                    // tile_dim1 | tile_dim2=0
  g1[5] = (int)stride0;                              // tensor_dim0_stride lo
  g1[6] = 0;                                         // stride0 hi | stride1 lo
  g1[7] = 0;

  i32x4 gz4 = {0, 0, 0, 0};
  i32x8 gz8 = {0, 0, 0, 0, 0, 0, 0, 0};
  __builtin_amdgcn_tensor_load_to_lds(g0, g1, gz4, gz4, gz8, 0);
}

// ---------------------------------------------------------------------------
// Kernel 1: projection GEMM  out[h][s][d] = bf16( (X @ W + b) * scale )
//   X: [S,E] f32, W: [E,E] f32, b: [E] f32 -> out: [H][S][D] bf16
// One WG computes a 64x64 output tile; 8 waves, each wave does 16(M)x32(N).
// ---------------------------------------------------------------------------
__global__ __launch_bounds__(256)
void qkv_gemm_kernel(const float* __restrict__ X,
                     const float* __restrict__ Wm,
                     const float* __restrict__ bias,
                     bf16* __restrict__ out,
                     float scale) {
  __shared__ bf16 sA[64][32];   // [m][k] row-major
  __shared__ bf16 sB[64][32];   // [n][k] (transposed for B fragments)

  const int tid  = threadIdx.x;
  const int wave = tid >> 5;
  const int lane = tid & 31;
  const int ln   = lane & 15;
  const int g    = lane >> 4;

  const int m0 = blockIdx.x * 64;
  const int n0 = blockIdx.y * 64;
  const int mw = (wave & 3) * 16;    // wave's M offset in tile
  const int nw = (wave >> 2) * 32;   // wave's N offset in tile

  v8f acc0 = {};
  v8f acc1 = {};

  for (int k0 = 0; k0 < E_DIM; k0 += 32) {
    // Prefetch next K-tile (lowers to global_prefetch_b8)
    if (k0 + 32 < E_DIM) {
      __builtin_prefetch(&X[(size_t)(m0 + (tid & 63)) * E_DIM + k0 + 32], 0, 0);
      __builtin_prefetch(&Wm[(size_t)(k0 + 32 + (tid & 31)) * E_DIM + n0 + ((tid >> 5) & 1) * 32], 0, 0);
    }
    // Stage A (64x32) fp32->bf16, vectorized float4 reads
    for (int idx = tid; idx < 64 * 32 / 4; idx += 256) {
      int r = idx >> 3, c = (idx & 7) * 4;
      const float4 f = *(const float4*)&X[(size_t)(m0 + r) * E_DIM + k0 + c];
      sA[r][c + 0] = (bf16)f.x; sA[r][c + 1] = (bf16)f.y;
      sA[r][c + 2] = (bf16)f.z; sA[r][c + 3] = (bf16)f.w;
    }
    // Stage B transposed: sB[n][c] = W[k0+c][n0+n]
    for (int idx = tid; idx < 64 * 32 / 4; idx += 256) {
      int c = idx >> 4, n4 = (idx & 15) * 4;
      const float4 f = *(const float4*)&Wm[(size_t)(k0 + c) * E_DIM + n0 + n4];
      sB[n4 + 0][c] = (bf16)f.x; sB[n4 + 1][c] = (bf16)f.y;
      sB[n4 + 2][c] = (bf16)f.z; sB[n4 + 3][c] = (bf16)f.w;
    }
    __syncthreads();

    // A fragment: lane row = mw+ln; elems k = g*16 + e (contiguous)
    v16bf a;
    {
      const int row = mw + ln;
      v8bf lo = *(const v8bf*)&sA[row][g * 16];
      v8bf hi = *(const v8bf*)&sA[row][g * 16 + 8];
#pragma unroll
      for (int e = 0; e < 8; ++e) { a[e] = lo[e]; a[e + 8] = hi[e]; }
    }
    // B fragments: lane col = ln(+16); elems k = g*16 + e (contiguous)
    v16bf b0 = *(const v16bf*)&sB[nw + ln][g * 16];
    v16bf b1 = *(const v16bf*)&sB[nw + 16 + ln][g * 16];

    acc0 = wmma_bf16(a, b0, acc0);
    acc1 = wmma_bf16(a, b1, acc1);
    __syncthreads();
  }

  // Epilogue: +bias, *scale, convert to bf16, scatter into head-major layout.
#pragma unroll
  for (int f = 0; f < 2; ++f) {
    v8f acc = f ? acc1 : acc0;
    const int n = n0 + nw + f * 16 + ln;
    const float bv = bias[n];
    const int h = n >> 6, d = n & 63;
#pragma unroll
    for (int r = 0; r < 8; ++r) {
      const int srow = m0 + mw + g * 8 + r;   // C layout: M = r + 8*laneGroup
      const float val = (acc[r] + bv) * scale;
      out[((size_t)h * S_LEN + srow) * D_DIM + d] = (bf16)val;
    }
  }
}

// ---------------------------------------------------------------------------
// Kernel 2: fused banded attention for one (head, 32-query block).
//   K chunks double-buffered through the Tensor Data Mover (TENSORcnt),
//   scores -> LDS strip, softmax in LDS, P@V -> d_out columns of this head.
// ---------------------------------------------------------------------------
__global__ __launch_bounds__(256)
void band_attn_kernel(const bf16* __restrict__ Q,
                      const bf16* __restrict__ K,
                      const bf16* __restrict__ V,
                      const float* __restrict__ attn_mask,
                      const unsigned char* __restrict__ idx_masked,
                      float* __restrict__ out) {
  __shared__ bf16  sQ[QB][D_DIM];      // 4 KB
  __shared__ bf16  sK[2][CH][D_DIM];   // 16 KB double-buffered K chunks [j][d]
  __shared__ bf16  sVT[D_DIM][CH];     // 8 KB V^T chunk [d][t]
  __shared__ float sS[QB][WN];         // 72 KB score / prob strip
  __shared__ float sRed[QB][8];        // softmax partials

  const int h  = blockIdx.y;
  const int q0 = blockIdx.x * QB;
  const int kstart = q0 - W_HALF;

  const int tid  = threadIdx.x;
  const int wave = tid >> 5;
  const int lane = tid & 31;
  const int ln   = lane & 15;
  const int g    = lane >> 4;

  // Wave-uniform (SGPR) wave index for TDM descriptor building.
  const int wuni = __builtin_amdgcn_readfirstlane(wave);
  const int t0   = wuni * 8;  // this wave's 8-row slice of each 64-row chunk

  // Load Q tile (vectorized 16B copies)
  for (int idx = tid; idx < QB * D_DIM / 8; idx += 256) {
    int r = idx >> 3, seg = (idx & 7) * 8;
    *(uint4*)&sQ[r][seg] =
        *(const uint4*)&Q[((size_t)h * S_LEN + q0 + r) * D_DIM + seg];
  }

  // ---- Pass 1: scores = Q . K^T over the window (double-buffered TDM) ----
  // Out-of-range key rows are clamped into [0, S-8]; those columns are
  // force-masked to -1e30 below, so their (garbage) K values never matter.
  const int mw = (wave & 1) * 16;    // score tile M base
  const int cw = (wave >> 1) * 16;   // score tile column base within chunk

  {  // prologue: chunk 0 into buffer 0
    int j0 = kstart + t0;
    int jc = j0 < 0 ? 0 : (j0 > (S_LEN - 8) ? (S_LEN - 8) : j0);
    tdm_load_rows_bf16(&K[((size_t)h * S_LEN + jc) * D_DIM],
                       (unsigned)(uintptr_t)&sK[0][t0][0], 8);
  }

  for (int cc = 0; cc < NCH; ++cc) {
    const int cbase = cc * CH;
    const int cur = cc & 1;

    if (cc + 1 < NCH) {
      // Issue next chunk into the spare buffer (its previous consumers all
      // passed the barrier at the end of the previous iteration), then wait
      // for the *current* chunk: per-wave TENSORcnt retires in order, so
      // tensorcnt<=1 means the older DMA has landed.
      int j0 = kstart + (cc + 1) * CH + t0;
      int jc = j0 < 0 ? 0 : (j0 > (S_LEN - 8) ? (S_LEN - 8) : j0);
      tdm_load_rows_bf16(&K[((size_t)h * S_LEN + jc) * D_DIM],
                         (unsigned)(uintptr_t)&sK[cur ^ 1][t0][0], 8);
      __builtin_amdgcn_s_wait_tensorcnt(1);
    } else {
      __builtin_amdgcn_s_wait_tensorcnt(0);
    }
    __syncthreads();  // all waves' slices of chunk cc visible (Q too, iter 0)

    v8f acc = {};
#pragma unroll
    for (int ks = 0; ks < 2; ++ks) {
      const int kb = ks * 32;
      v16bf a;
      {
        const int row = mw + ln;
        v8bf lo = *(const v8bf*)&sQ[row][kb + g * 16];
        v8bf hi = *(const v8bf*)&sQ[row][kb + g * 16 + 8];
#pragma unroll
        for (int e = 0; e < 8; ++e) { a[e] = lo[e]; a[e + 8] = hi[e]; }
      }
      v16bf b = *(const v16bf*)&sK[cur][cw + ln][kb + g * 16];
      acc = wmma_bf16(a, b, acc);
    }

    // Mask and write into the score strip
    const int cwin = cbase + cw + ln;
    const int j = kstart + cwin;
    const bool jin = (j >= 0) && (j < S_LEN);
    const float km = (jin && attn_mask[jin ? j : 0] != 0.0f) ? LARGE_NEG : 0.0f;
#pragma unroll
    for (int r = 0; r < 8; ++r) {
      const int m = mw + g * 8 + r;
      const bool valid = jin && (cwin >= m) && (cwin <= m + 2 * W_HALF);
      sS[m][cwin] = valid ? (acc[r] + km) : -1.0e30f;
    }
    __syncthreads();  // chunk-cc buffer free for the TDM two iterations later
  }

  // ---- Softmax over each 576-wide row (8 threads per row) ----
  {
    const int m = tid >> 3;
    const int p = tid & 7;
    const int c0 = p * (WN / 8);
    float mx = -1.0e30f;
    for (int c = c0; c < c0 + WN / 8; ++c) mx = fmaxf(mx, sS[m][c]);
    sRed[m][p] = mx;
    __syncthreads();
#pragma unroll
    for (int q = 0; q < 8; ++q) mx = fmaxf(mx, sRed[m][q]);
    float sum = 0.0f;
    for (int c = c0; c < c0 + WN / 8; ++c) {
      float e = __expf(sS[m][c] - mx);
      sS[m][c] = e;
      sum += e;
    }
    __syncthreads();
    sRed[m][p] = sum;
    __syncthreads();
    sum = 0.0f;
#pragma unroll
    for (int q = 0; q < 8; ++q) sum += sRed[m][q];
    float inv = 1.0f / sum;
    if (idx_masked[q0 + m]) inv = 0.0f;   // is_index_masked zeroes probs
    for (int c = c0; c < c0 + WN / 8; ++c) sS[m][c] *= inv;
  }
  __syncthreads();

  // ---- Pass 2: O = P @ V ----
  const int mw2 = (wave & 1) * 16;
  const int nw2 = (wave >> 1) * 16;   // head-dim column base
  v8f o = {};
  for (int cc = 0; cc < NCH; ++cc) {
    const int cbase = cc * CH;
    // Stage V chunk transposed: sVT[d][t]  (probs at OOB rows are zero)
    for (int idx = tid; idx < CH * D_DIM; idx += 256) {
      int t = idx >> 6, d = idx & 63;
      int j = kstart + cbase + t;
      sVT[d][t] = (j >= 0 && j < S_LEN)
                      ? V[((size_t)h * S_LEN + j) * D_DIM + d]
                      : (bf16)0.0f;
    }
    __syncthreads();

#pragma unroll
    for (int ks = 0; ks < 2; ++ks) {
      const int kb = ks * 32;
      v16bf a;
      {
        const int row = mw2 + ln;
#pragma unroll
        for (int e = 0; e < 8; ++e) {
          a[e]     = (bf16)sS[row][cbase + kb + g * 16 + e];
          a[e + 8] = (bf16)sS[row][cbase + kb + g * 16 + 8 + e];
        }
      }
      v16bf b = *(const v16bf*)&sVT[nw2 + ln][kb + g * 16];
      o = wmma_bf16(a, b, o);
    }
    __syncthreads();  // sVT free for next chunk
  }

  // Write 32x64 f32 block into this head's columns of the output
  {
    const int col = h * D_DIM + nw2 + ln;
#pragma unroll
    for (int r = 0; r < 8; ++r) {
      const int i = q0 + mw2 + g * 8 + r;
      out[(size_t)i * E_DIM + col] = o[r];
    }
  }
}

// ---------------------------------------------------------------------------
// Host launcher
// ---------------------------------------------------------------------------
extern "C" void kernel_launch(void* const* d_in, const int* in_sizes, int n_in,
                              void* d_out, int out_size, void* d_ws, size_t ws_size,
                              hipStream_t stream) {
  (void)in_sizes; (void)n_in; (void)out_size; (void)ws_size;

  const float* hs = (const float*)d_in[0];                 // (S,E)
  const float* am = (const float*)d_in[1];                 // (S,)
  const unsigned char* im = (const unsigned char*)d_in[2]; // (S,) bool
  const float* Wq = (const float*)d_in[3];
  const float* bq = (const float*)d_in[4];
  const float* Wk = (const float*)d_in[5];
  const float* bk = (const float*)d_in[6];
  const float* Wv = (const float*)d_in[7];
  const float* bv = (const float*)d_in[8];
  float* out = (float*)d_out;

  const size_t HSD = (size_t)H_NUM * S_LEN * D_DIM;
  bf16* Qb = (bf16*)d_ws;
  bf16* Kb = Qb + HSD;
  bf16* Vb = Kb + HSD;

  dim3 gg(S_LEN / 64, E_DIM / 64);
  qkv_gemm_kernel<<<gg, 256, 0, stream>>>(hs, Wq, bq, Qb, 0.125f); // 1/sqrt(64)
  qkv_gemm_kernel<<<gg, 256, 0, stream>>>(hs, Wk, bk, Kb, 1.0f);
  qkv_gemm_kernel<<<gg, 256, 0, stream>>>(hs, Wv, bv, Vb, 1.0f);

  dim3 ga(S_LEN / QB, H_NUM);
  band_attn_kernel<<<ga, 256, 0, stream>>>(Qb, Kb, Vb, am, im, out);
}
